// DeformableNet_26766236188676
// MI455X (gfx1250) — matrix operand order repaired and checked by reading, not compiled, not run on hardware
//
#include <hip/hip_runtime.h>

typedef _Float16 v16h __attribute__((ext_vector_type(16)));
typedef _Float16 half8 __attribute__((ext_vector_type(8)));
typedef float v8f __attribute__((ext_vector_type(8)));

#define ALPHA 0.2f
#define BN_EPS 0.001f

__device__ __forceinline__ float leaky_f(float x) { return x >= 0.f ? x : ALPHA * x; }

// ---------------------------------------------------------------------------
// Stage 0: conv3x3 (Cin=2: fixed,moving) -> 32ch, +bias, leaky, 2x2 avgpool, BN
// All 9 taps x 2 cin packed into WMMA K dimension (K=18, zero padded to 32).
// Grid: (W/16, H/8, B), block 256 (8 waves, one 16-px row per wave).
// ---------------------------------------------------------------------------
__global__ __launch_bounds__(256) void stage0_kernel(
    const float* __restrict__ fixedp, const float* __restrict__ movingp,
    const float* __restrict__ w, const float* __restrict__ bias,
    const float* __restrict__ g, const float* __restrict__ be,
    const float* __restrict__ mn, const float* __restrict__ vr,
    _Float16* __restrict__ out)
{
  const int H = 512, W = 512;
  __shared__ __align__(16) _Float16 wl[32][32];   // [n=cout][k=tap*2+cin], k>=18 -> 0
  __shared__ __align__(16) _Float16 pl[10][18][2];
  __shared__ __align__(16) float    cl[8][16][32];
  const int tid = threadIdx.x;
  const int x0 = blockIdx.x * 16, y0 = blockIdx.y * 8, b = blockIdx.z;

  // weights (3,3,2,32) HWIO: elem(tap,cin,cout) at k*32+n with k=tap*2+cin
  for (int idx = tid; idx < 1024; idx += 256) {
    int n = idx >> 5, k = idx & 31;
    wl[n][k] = (k < 18) ? (_Float16)w[k * 32 + n] : (_Float16)0.f;
  }
  // input patch rows y0-1..y0+8, cols x0-1..x0+16, zero pad (SAME)
  for (int idx = tid; idx < 180; idx += 256) {
    int i = idx / 18, j = idx % 18;
    int gy = y0 - 1 + i, gx = x0 - 1 + j;
    bool ok = (gy >= 0 && gy < H && gx >= 0 && gx < W);
    size_t o = ((size_t)b * H + gy) * W + gx;
    pl[i][j][0] = ok ? (_Float16)fixedp[o]  : (_Float16)0.f;
    pl[i][j][1] = ok ? (_Float16)movingp[o] : (_Float16)0.f;
  }
  __syncthreads();

  const int wave = tid >> 5, lane = tid & 31;
  const int m = lane & 15, hi = lane >> 4;

  // A fragment: m = pixel in tile, k = tap*2+cin (ISA wave32 16-bit A layout)
  v16h a;
#pragma unroll
  for (int e = 0; e < 16; ++e) {
    int k = (e & 7) + 8 * hi + ((e >= 8) ? 16 : 0);
    _Float16 val = (_Float16)0.f;
    if (k < 18) {
      int tap = k >> 1, cin = k & 1;
      int dy = tap / 3, dx = tap % 3;
      val = pl[wave + dy][m + dx][cin];
    }
    a[e] = val;
  }
  const int kb = 8 * hi;
  v16h b0, b1;
  ((half8*)&b0)[0] = *(const half8*)&wl[m][kb];
  ((half8*)&b0)[1] = *(const half8*)&wl[m][kb + 16];
  ((half8*)&b1)[0] = *(const half8*)&wl[m + 16][kb];
  ((half8*)&b1)[1] = *(const half8*)&wl[m + 16][kb + 16];
  v8f c0 = {}, c1 = {};
  c0 = __builtin_amdgcn_wmma_f32_16x16x32_f16(false, a, false, b0, (short)0, c0, false, false);
  c1 = __builtin_amdgcn_wmma_f32_16x16x32_f16(false, a, false, b1, (short)0, c1, false, false);

  // bias + leaky -> LDS (C/D layout: n = lane&15, m = r + 8*hi)
#pragma unroll
  for (int r = 0; r < 8; ++r) {
    int mm = r + 8 * hi;
    cl[wave][mm][m]      = leaky_f(c0[r] + bias[m]);
    cl[wave][mm][m + 16] = leaky_f(c1[r] + bias[m + 16]);
  }
  __syncthreads();

  // 2x2 avgpool + BN -> fp16 out at 256x256
  const int Ho = 256, Wo = 256;
  for (int idx = tid; idx < 1024; idx += 256) {
    int ch = idx & 31, pc = (idx >> 5) & 7, pr = idx >> 8;
    float s = cl[2*pr][2*pc][ch] + cl[2*pr][2*pc+1][ch]
            + cl[2*pr+1][2*pc][ch] + cl[2*pr+1][2*pc+1][ch];
    float v = 0.25f * s;
    v = (v - mn[ch]) * rsqrtf(vr[ch] + BN_EPS) * g[ch] + be[ch];
    out[(((size_t)b * Ho + ((y0 >> 1) + pr)) * Wo + ((x0 >> 1) + pc)) * 32 + ch] = (_Float16)v;
  }
}

// ---------------------------------------------------------------------------
// Generic conv stage: Cin=32 -> Cout=32, KS in {1,3}, optional pool+BN fusion.
// Grid: (W/16, H/8, B), block 256.
// ---------------------------------------------------------------------------
template <int KS, int POOL>
__global__ __launch_bounds__(256) void conv_stage(
    const _Float16* __restrict__ in, int H, int W,
    const float* __restrict__ w, const float* __restrict__ bias,
    const float* __restrict__ g, const float* __restrict__ be,
    const float* __restrict__ mn, const float* __restrict__ vr,
    _Float16* __restrict__ out)
{
  constexpr int NT  = KS * KS;
  constexpr int PR  = 8 + KS - 1;
  constexpr int PC  = 16 + KS - 1;
  constexpr int PAD = KS / 2;
  __shared__ __align__(16) _Float16 wl[NT][32][32];  // [tap][n][k]
  __shared__ __align__(16) _Float16 al[PR][PC][32];  // input patch (halo)
  __shared__ __align__(16) float    cl[8][16][32];   // conv output (for pool)
  const int tid = threadIdx.x;
  const int x0 = blockIdx.x * 16, y0 = blockIdx.y * 8, b = blockIdx.z;

  // weights HWIO [tap][k=cin][n=cout] -> LDS transposed [tap][n][k]
  for (int idx = tid; idx < NT * 1024; idx += 256) {
    int tap = idx >> 10, r = idx & 1023;
    int k = r >> 5, n = r & 31;
    wl[tap][n][k] = (_Float16)w[(tap * 32 + k) * 32 + n];
  }
  // activation patch with zero padding, 8-half vector chunks
  for (int cidx = tid; cidx < PR * PC * 4; cidx += 256) {
    int pix = cidx >> 2, cc = (cidx & 3) * 8;
    int i = pix / PC, j = pix % PC;
    int gy = y0 - PAD + i, gx = x0 - PAD + j;
    half8 v = {};
    if (gy >= 0 && gy < H && gx >= 0 && gx < W)
      v = *(const half8*)&in[(((size_t)b * H + gy) * W + gx) * 32 + cc];
    *(half8*)&al[i][j][cc] = v;
  }
  __syncthreads();

  const int wave = tid >> 5, lane = tid & 31;
  const int n0 = lane & 15, hi = lane >> 4;  // n0 doubles as A's pixel index m
  const int cb = 8 * hi;                     // A channel base (k layout)
  const int kb = 8 * hi;                     // B k base

  v8f c0 = {}, c1 = {};
#pragma unroll
  for (int tap = 0; tap < NT; ++tap) {
    const int dy = tap / KS, dx = tap % KS;
    v16h a, bf0, bf1;
    // A: elements 0..7 -> channels cb..cb+7, elements 8..15 -> cb+16..cb+23
    ((half8*)&a)[0] = *(const half8*)&al[wave + dy][n0 + dx][cb];
    ((half8*)&a)[1] = *(const half8*)&al[wave + dy][n0 + dx][cb + 16];
    // B: k values kb..kb+7 and kb+16..kb+23 for output channel n (two N-halves)
    ((half8*)&bf0)[0] = *(const half8*)&wl[tap][n0][kb];
    ((half8*)&bf0)[1] = *(const half8*)&wl[tap][n0][kb + 16];
    ((half8*)&bf1)[0] = *(const half8*)&wl[tap][n0 + 16][kb];
    ((half8*)&bf1)[1] = *(const half8*)&wl[tap][n0 + 16][kb + 16];
    c0 = __builtin_amdgcn_wmma_f32_16x16x32_f16(false, a, false, bf0, (short)0, c0, false, false);
    c1 = __builtin_amdgcn_wmma_f32_16x16x32_f16(false, a, false, bf1, (short)0, c1, false, false);
  }

  if (POOL) {
#pragma unroll
    for (int r = 0; r < 8; ++r) {
      int mm = r + 8 * hi;
      cl[wave][mm][n0]      = leaky_f(c0[r] + bias[n0]);
      cl[wave][mm][n0 + 16] = leaky_f(c1[r] + bias[n0 + 16]);
    }
    __syncthreads();
    const int Ho = H >> 1, Wo = W >> 1;
    for (int idx = tid; idx < 1024; idx += 256) {
      int ch = idx & 31, pc = (idx >> 5) & 7, pr = idx >> 8;
      float s = cl[2*pr][2*pc][ch] + cl[2*pr][2*pc+1][ch]
              + cl[2*pr+1][2*pc][ch] + cl[2*pr+1][2*pc+1][ch];
      float v = 0.25f * s;
      v = (v - mn[ch]) * rsqrtf(vr[ch] + BN_EPS) * g[ch] + be[ch];
      out[(((size_t)b * Ho + ((y0 >> 1) + pr)) * Wo + ((x0 >> 1) + pc)) * 32 + ch] = (_Float16)v;
    }
  } else {
    const float rs0 = rsqrtf(vr[n0] + BN_EPS) * g[n0];
    const float rs1 = rsqrtf(vr[n0 + 16] + BN_EPS) * g[n0 + 16];
#pragma unroll
    for (int r = 0; r < 8; ++r) {
      int mm = r + 8 * hi;
      float v0 = leaky_f(c0[r] + bias[n0]);
      float v1 = leaky_f(c1[r] + bias[n0 + 16]);
      v0 = (v0 - mn[n0])      * rs0 + be[n0];
      v1 = (v1 - mn[n0 + 16]) * rs1 + be[n0 + 16];
      size_t o = (((size_t)b * H + (y0 + wave)) * W + (x0 + mm)) * 32;
      out[o + n0]      = (_Float16)v0;
      out[o + n0 + 16] = (_Float16)v1;
    }
  }
}

// ---------------------------------------------------------------------------
// Final 1x1 conv: 32 -> 2, fp32 disp out. 8*16*16 = 2048 pixels.
// ---------------------------------------------------------------------------
__global__ __launch_bounds__(256) void final_conv(
    const _Float16* __restrict__ in, const float* __restrict__ w,
    const float* __restrict__ bias, float* __restrict__ disp)
{
  int px = blockIdx.x * 256 + threadIdx.x;
  if (px >= 8 * 16 * 16) return;
  float a0 = bias[0], a1 = bias[1];
#pragma unroll
  for (int k = 0; k < 32; ++k) {
    float x = (float)in[(size_t)px * 32 + k];
    a0 += x * w[k * 2 + 0];
    a1 += x * w[k * 2 + 1];
  }
  disp[px * 2 + 0] = a0;
  disp[px * 2 + 1] = a1;
}

// ---------------------------------------------------------------------------
// Cubic B-spline upsample of disp (16x16x2 grid, symmetric pad) + bilinear warp.
// Grid: (512*512/256, B), block 256.
// ---------------------------------------------------------------------------
__device__ __forceinline__ int mir16(int q) {
  q = q < 0 ? -1 - q : q;
  return q > 15 ? 31 - q : q;
}

__global__ __launch_bounds__(256) void warp_kernel(
    const float* __restrict__ disp, const float* __restrict__ fixedp,
    float* __restrict__ outp)
{
  __shared__ float dl[16 * 16 * 2];
  const int b = blockIdx.y;
  const int tid = threadIdx.x;
  dl[tid]       = disp[b * 512 + tid];
  dl[tid + 256] = disp[b * 512 + tid + 256];
  __syncthreads();

  const int p = blockIdx.x * 256 + tid;
  const int y = p >> 9, x = p & 511;
  const float S = 15.0f / 511.0f;
  float fx = x * S, fy = y * S;
  float u = fx * (1.0f / 16.0f), vv = fy * (1.0f / 16.0f);
  int i = (int)floorf(fx), j = (int)floorf(fy);
  float u2 = u * u, u3 = u2 * u;
  float v2 = vv * vv, v3 = v2 * vv;
  const float k6 = 1.0f / 6.0f;
  float Bu[4] = { (-u3 + 3.f*u2 - 3.f*u + 1.f) * k6,
                  ( 3.f*u3 - 6.f*u2 + 4.f) * k6,
                  (-3.f*u3 + 3.f*u2 + 3.f*u + 1.f) * k6,
                  u3 * k6 };
  float Bv[4] = { (-v3 + 3.f*v2 - 3.f*vv + 1.f) * k6,
                  ( 3.f*v3 - 6.f*v2 + 4.f) * k6,
                  (-3.f*v3 + 3.f*v2 + 3.f*vv + 1.f) * k6,
                  v3 * k6 };
  float d0 = 0.f, d1 = 0.f;
#pragma unroll
  for (int r = 0; r < 4; ++r) {
    int jq = mir16(j + r - 1);
#pragma unroll
    for (int s = 0; s < 4; ++s) {
      int iq = mir16(i + s - 1);
      float wgt = Bu[r] * Bv[s];    // reference einsum: Bu over row offs, Bv over col offs
      d0 += wgt * dl[(jq * 16 + iq) * 2 + 0];
      d1 += wgt * dl[(jq * 16 + iq) * 2 + 1];
    }
  }
  float x_t = d0 + (float)x, y_t = d1 + (float)y;
  float x0f = floorf(x_t), y0f = floorf(y_t);
  float x0 = fminf(fmaxf(x0f, 0.f), 511.f);
  float y0 = fminf(fmaxf(y0f, 0.f), 511.f);
  float x1 = fminf(fmaxf(x0f + 1.f, 0.f), 511.f);
  float y1 = fminf(fmaxf(y0f + 1.f, 0.f), 511.f);
  const float* im = fixedp + (size_t)b * 512 * 512;
  int ix0 = (int)x0, iy0 = (int)y0, ix1 = (int)x1, iy1 = (int)y1;
  float Q1 = im[iy0 * 512 + ix0];
  float Q2 = im[iy1 * 512 + ix0];
  float Q3 = im[iy0 * 512 + ix1];
  float Q4 = im[iy1 * 512 + ix1];
  const float eps = 1e-5f;
  float ddx = x1 - x0 + eps, ddy = y1 - y0 + eps;
  float wx1 = (x1 - x_t) / ddx, wx0 = (x_t - x0) / ddx;
  float R1 = wx1 * Q1 + wx0 * Q3;
  float R2 = wx1 * Q2 + wx0 * Q4;
  float o = (y1 - y_t) / ddy * R1 + (y_t - y0) / ddy * R2;
  outp[(size_t)b * 512 * 512 + p] = o;
}

// ---------------------------------------------------------------------------
extern "C" void kernel_launch(void* const* d_in, const int* in_sizes, int n_in,
                              void* d_out, int out_size, void* d_ws, size_t ws_size,
                              hipStream_t stream) {
  (void)in_sizes; (void)n_in; (void)out_size; (void)ws_size;
  const float* fixedp  = (const float*)d_in[0];
  const float* movingp = (const float*)d_in[1];
  const float* W[10]; const float* Bi[10];
  for (int i = 0; i < 10; ++i) { W[i] = (const float*)d_in[2 + i]; Bi[i] = (const float*)d_in[12 + i]; }
  const float *G[9], *Be[9], *Mn[9], *Vr[9];
  for (int i = 0; i < 9; ++i) {
    G[i]  = (const float*)d_in[22 + i];
    Be[i] = (const float*)d_in[31 + i];
    Mn[i] = (const float*)d_in[40 + i];
    Vr[i] = (const float*)d_in[49 + i];
  }
  char* ws = (char*)d_ws;
  _Float16* actA = (_Float16*)ws;                                  // <= 33.6 MB
  _Float16* actB = (_Float16*)(ws + (size_t)34 * 1024 * 1024);     // <= 8.4 MB
  float*    disp = (float*)(ws + (size_t)43 * 1024 * 1024);        // 16 KB

  // 5 pooled stages: 512 -> 256 -> 128 -> 64 -> 32 -> 16
  stage0_kernel<<<dim3(32, 64, 8), 256, 0, stream>>>(fixedp, movingp, W[0], Bi[0],
                                                     G[0], Be[0], Mn[0], Vr[0], actA);
  conv_stage<3, 1><<<dim3(16, 32, 8), 256, 0, stream>>>(actA, 256, 256, W[1], Bi[1],
                                                        G[1], Be[1], Mn[1], Vr[1], actB);
  conv_stage<3, 1><<<dim3(8, 16, 8), 256, 0, stream>>>(actB, 128, 128, W[2], Bi[2],
                                                       G[2], Be[2], Mn[2], Vr[2], actA);
  conv_stage<3, 1><<<dim3(4, 8, 8), 256, 0, stream>>>(actA, 64, 64, W[3], Bi[3],
                                                      G[3], Be[3], Mn[3], Vr[3], actB);
  conv_stage<3, 1><<<dim3(2, 4, 8), 256, 0, stream>>>(actB, 32, 32, W[4], Bi[4],
                                                      G[4], Be[4], Mn[4], Vr[4], actA);
  // head at 16x16: conv3x3, conv3x3, conv1x1, conv1x1 (leaky+BN each)
  conv_stage<3, 0><<<dim3(1, 2, 8), 256, 0, stream>>>(actA, 16, 16, W[5], Bi[5],
                                                      G[5], Be[5], Mn[5], Vr[5], actB);
  conv_stage<3, 0><<<dim3(1, 2, 8), 256, 0, stream>>>(actB, 16, 16, W[6], Bi[6],
                                                      G[6], Be[6], Mn[6], Vr[6], actA);
  conv_stage<1, 0><<<dim3(1, 2, 8), 256, 0, stream>>>(actA, 16, 16, W[7], Bi[7],
                                                      G[7], Be[7], Mn[7], Vr[7], actB);
  conv_stage<1, 0><<<dim3(1, 2, 8), 256, 0, stream>>>(actB, 16, 16, W[8], Bi[8],
                                                      G[8], Be[8], Mn[8], Vr[8], actA);
  // final 1x1 -> 2ch disp
  final_conv<<<dim3(8), 256, 0, stream>>>(actA, W[9], Bi[9], disp);
  // B-spline upsample + bilinear warp
  warp_kernel<<<dim3(1024, 8), 256, 0, stream>>>(disp, fixedp, (float*)d_out);
}
